// SelfAttention_56968446214802
// MI455X (gfx1250) — compile-verified, matrix operand order checked by reading, not run on hardware
//
#include <hip/hip_runtime.h>

// Deformable self-attention (MSDeformAttn) for MI455X / gfx1250.
// BS=8, Nq=128*128=16384, C=128, H=8, P=4, hd=16, R=128.
//
// Stage 0: pack weights into per-lane WMMA B-fragment layout (f16)
// Stage 1: fused projection  query @ [Wv|Woff|Wattn]   (f16 WMMA, f32 acc)
//          A staged in LDS already permuted to fragment order -> wide ds loads
// Stage 2: softmax over P + bilinear sampling + P-reduction; output written
//          directly in stage-3 A-fragment layout (f16)
// Stage 3: out = sampled @ Wout + bout + 2*query  (f16 WMMA, no LDS, no barrier)

#define NQ   16384
#define RR   128
#define NTOT 224          // 128 (value) + 64 (offsets) + 32 (attn logits)

typedef __attribute__((ext_vector_type(16))) _Float16 v16h;
typedef __attribute__((ext_vector_type(8)))  _Float16 v8h;
typedef __attribute__((ext_vector_type(8)))  float    v8f;

#define WMMA_F16(A, B, C) \
    __builtin_amdgcn_wmma_f32_16x16x32_f16(false, (A), false, (B), (short)0, (C), false, false)

// K-index of element e of the per-lane 16-half fragment (ISA 7.12.2,
// "16-bit A-Matrix 16x32"): VGPR 0-3 -> K=0..7 (+8*khalf), VGPR 4-7 -> K=16..23 (+8*khalf).
__device__ __forceinline__ int kmap16(int e) {
    int v  = e >> 1;
    int k0 = (v < 4) ? (2 * v) : (16 + 2 * (v - 4));
    return k0 + (e & 1);
}

// Inverse of the fragment K-map: channel-in-chunk kin (0..31) -> element index
__device__ __forceinline__ int inv_e(int kin) {
    return kin - ((kin >= 8) ? 8 : 0) - ((kin >= 24) ? 8 : 0);
}

// ---------------------------------------------------------------- stage 0: pack B fragments
// BallP: [14 tiles][4 kk][32 lanes][16 halfs];  BoutP: [8 tiles][4 kk][32 lanes][16 halfs]
__global__ __launch_bounds__(256) void prep_pack(const float* __restrict__ Wv,
                                                 const float* __restrict__ Woff,
                                                 const float* __restrict__ Wattn,
                                                 const float* __restrict__ Wout,
                                                 _Float16* __restrict__ BallP,
                                                 _Float16* __restrict__ BoutP) {
    int idx = blockIdx.x * 256 + threadIdx.x;
    if (idx < 14 * 4 * 32 * 16) {
        int e = idx & 15, lane = (idx >> 4) & 31, kk = (idx >> 9) & 3, tile = idx >> 11;
        int n = tile * 16 + (lane & 15);
        int k = kk * 32 + (lane >> 4) * 8 + kmap16(e);
        float w = (n < 128) ? Wv[k * 128 + n]
                 : (n < 192) ? Woff[k * 64 + (n - 128)]
                             : Wattn[k * 32 + (n - 192)];
        BallP[idx] = (_Float16)w;
    }
    if (idx < 8 * 4 * 32 * 16) {
        int e = idx & 15, lane = (idx >> 4) & 31, kk = (idx >> 9) & 3, tile = idx >> 11;
        int n = tile * 16 + (lane & 15);
        int k = kk * 32 + (lane >> 4) * 8 + kmap16(e);
        BoutP[idx] = (_Float16)Wout[k * 128 + n];
    }
}

// ---------------------------------------------------------------- stage 1: projections
// grid = BS * (NQ/16) = 8192 blocks, 256 threads (8 waves). Each block: one 16-row
// query tile; wave w computes N-tiles {w, w+8} of the 224-wide fused output.
__global__ __launch_bounds__(256) void msda_proj(const float* __restrict__ query,
                                                 const _Float16* __restrict__ BallP,
                                                 const float* __restrict__ bv,
                                                 const float* __restrict__ boff,
                                                 const float* __restrict__ battn,
                                                 float* __restrict__ valueB,   // (b,h,q,16)
                                                 float* __restrict__ offB,     // (b,q,64)
                                                 float* __restrict__ logitB) { // (b,q,32)
    __shared__ _Float16 lq[4 * 32 * 16];   // A fragments: [kk][lane][16], 4 KB
    int b     = blockIdx.x >> 10;
    int mtile = blockIdx.x & 1023;
    int q0    = mtile * 16;

    const float* qrow = query + ((size_t)b * NQ + q0) * 128;
    for (int i = threadIdx.x; i < 16 * 128; i += 256) {
        int r = i >> 7, k = i & 127;
        int kk = k >> 5, kin = k & 31;
        int khalf = (kin >> 3) & 1;
        lq[((kk * 32 + r + 16 * khalf) << 4) + inv_e(kin)] = (_Float16)qrow[i];
    }
    __syncthreads();

    int w = threadIdx.x >> 5, lane = threadIdx.x & 31;
    int col = lane & 15, khalf = lane >> 4;   // N column and K-half for this lane

    // A fragments for this lane, held in VGPRs across both N-tiles.
    v16h a0 = *reinterpret_cast<const v16h*>(&lq[(0 * 32 + lane) << 4]);
    v16h a1 = *reinterpret_cast<const v16h*>(&lq[(1 * 32 + lane) << 4]);
    v16h a2 = *reinterpret_cast<const v16h*>(&lq[(2 * 32 + lane) << 4]);
    v16h a3 = *reinterpret_cast<const v16h*>(&lq[(3 * 32 + lane) << 4]);

    for (int ntile = w; ntile < NTOT / 16; ntile += 8) {   // wave-uniform
        // Issue all B-fragment loads up front, then run the WMMA chain.
        const _Float16* bp = BallP + ((size_t)ntile * 4 * 32 + lane) * 16;
        v16h b0 = *reinterpret_cast<const v16h*>(bp);
        v16h b1 = *reinterpret_cast<const v16h*>(bp + 32 * 16);
        v16h b2 = *reinterpret_cast<const v16h*>(bp + 2 * 32 * 16);
        v16h b3 = *reinterpret_cast<const v16h*>(bp + 3 * 32 * 16);

        v8f acc = {};
        acc = WMMA_F16(a0, b0, acc);
        acc = WMMA_F16(a1, b1, acc);
        acc = WMMA_F16(a2, b2, acc);
        acc = WMMA_F16(a3, b3, acc);

        // Tile-uniform epilogue: classify once, store with immediate offsets.
        int n = ntile * 16 + col;
        if (n < 128) {           // value, (b,h,q,hd) for contiguous 64B gathers
            float bias = bv[n];
            float* p = valueB + (((size_t)b * 8 + (n >> 4)) * NQ + q0 + khalf * 8) * 16 + (n & 15);
#pragma unroll
            for (int i = 0; i < 8; ++i) p[i * 16] = acc[i] + bias;
        } else if (n < 192) {    // sampling offsets
            float bias = boff[n - 128];
            float* p = offB + ((size_t)b * NQ + q0 + khalf * 8) * 64 + (n - 128);
#pragma unroll
            for (int i = 0; i < 8; ++i) p[i * 64] = acc[i] + bias;
        } else {                 // attention logits (softmax deferred)
            float bias = battn[n - 192];
            float* p = logitB + ((size_t)b * NQ + q0 + khalf * 8) * 32 + (n - 192);
#pragma unroll
            for (int i = 0; i < 8; ++i) p[i * 32] = acc[i] + bias;
        }
    }
}

// ---------------------------------------------------------------- stage 2: sampling
// One thread per (b,q,h): softmax over P=4, 4 points x 4 bilinear taps, each tap a
// contiguous 64B load of the 16-channel head slice. Output written straight into the
// stage-3 A-fragment layout: attnP[(b*1024+mtile)*2048 + (kk*32+lane)*16 + e].
__global__ __launch_bounds__(256) void msda_sample(const float* __restrict__ valueB,
                                                   const float* __restrict__ offB,
                                                   const float* __restrict__ logitB,
                                                   _Float16* __restrict__ attnP) {
    int tid = blockIdx.x * 256 + threadIdx.x;
    int h = tid & 7;
    int q = (tid >> 3) & (NQ - 1);
    int b = tid >> 17;
    size_t bq = (size_t)b * NQ + q;

    const float* lg = logitB + bq * 32 + h * 4;
    float l0 = lg[0], l1 = lg[1], l2 = lg[2], l3 = lg[3];
    float m  = fmaxf(fmaxf(l0, l1), fmaxf(l2, l3));
    float e0 = __expf(l0 - m), e1 = __expf(l1 - m), e2 = __expf(l2 - m), e3 = __expf(l3 - m);
    float inv = 1.0f / (e0 + e1 + e2 + e3);
    float aw[4] = {e0 * inv, e1 * inv, e2 * inv, e3 * inv};

    const float* of = offB + bq * 64 + h * 8;
    // x = ref_x*R - 0.5 + off_x, ref_x = (q%R)/(R-1)
    float bx = (float)(q & 127) * (128.0f / 127.0f) - 0.5f;
    float by = (float)(q >> 7) * (128.0f / 127.0f) - 0.5f;
    const float* vbase = valueB + ((size_t)(b * 8 + h) * NQ) * 16;

    float acc[16];
#pragma unroll
    for (int d = 0; d < 16; ++d) acc[d] = 0.0f;

#define MSDA_TAP(XI, YI, WT)                                                            \
    do {                                                                                \
        int xi_ = (XI), yi_ = (YI);                                                     \
        if (xi_ >= 0 && xi_ < RR && yi_ >= 0 && yi_ < RR) {                             \
            const float4* vp_ = reinterpret_cast<const float4*>(                        \
                vbase + ((size_t)(yi_ * RR + xi_)) * 16);                               \
            float wt_ = (WT);                                                           \
            float4 t0_ = vp_[0], t1_ = vp_[1], t2_ = vp_[2], t3_ = vp_[3];              \
            acc[0]  = fmaf(wt_, t0_.x, acc[0]);  acc[1]  = fmaf(wt_, t0_.y, acc[1]);    \
            acc[2]  = fmaf(wt_, t0_.z, acc[2]);  acc[3]  = fmaf(wt_, t0_.w, acc[3]);    \
            acc[4]  = fmaf(wt_, t1_.x, acc[4]);  acc[5]  = fmaf(wt_, t1_.y, acc[5]);    \
            acc[6]  = fmaf(wt_, t1_.z, acc[6]);  acc[7]  = fmaf(wt_, t1_.w, acc[7]);    \
            acc[8]  = fmaf(wt_, t2_.x, acc[8]);  acc[9]  = fmaf(wt_, t2_.y, acc[9]);    \
            acc[10] = fmaf(wt_, t2_.z, acc[10]); acc[11] = fmaf(wt_, t2_.w, acc[11]);   \
            acc[12] = fmaf(wt_, t3_.x, acc[12]); acc[13] = fmaf(wt_, t3_.y, acc[13]);   \
            acc[14] = fmaf(wt_, t3_.z, acc[14]); acc[15] = fmaf(wt_, t3_.w, acc[15]);   \
        }                                                                               \
    } while (0)

#pragma unroll
    for (int p = 0; p < 4; ++p) {
        float x = bx + of[2 * p];
        float y = by + of[2 * p + 1];
        float x0f = floorf(x), y0f = floorf(y);
        float fx = x - x0f, fy = y - y0f;
        int x0 = (int)x0f, y0 = (int)y0f;
        int x1 = x0 + 1, y1 = y0 + 1;
        MSDA_TAP(x0, y0, aw[p] * (1.0f - fx) * (1.0f - fy));
        MSDA_TAP(x1, y0, aw[p] * fx * (1.0f - fy));
        MSDA_TAP(x0, y1, aw[p] * (1.0f - fx) * fy);
        MSDA_TAP(x1, y1, aw[p] * fx * fy);
    }
#undef MSDA_TAP

    // Write channels c = h*16 + d of row q into fragment layout.
    // d=0..7  -> lane = (q&15),      element offset = (h&1)*8
    // d=8..15 -> lane = (q&15)+16,   element offset = (h&1)*8
    _Float16* base = attnP + ((size_t)b * 1024 + (q >> 4)) * 2048;
    int r = q & 15, kk = h >> 1, eoff = (h & 1) * 8;
    v8h lo, hi;
#pragma unroll
    for (int j = 0; j < 8; ++j) { lo[j] = (_Float16)acc[j]; hi[j] = (_Float16)acc[8 + j]; }
    *reinterpret_cast<v8h*>(base + ((kk * 32 + r) << 4) + eoff)      = lo;
    *reinterpret_cast<v8h*>(base + ((kk * 32 + r + 16) << 4) + eoff) = hi;
}

// ---------------------------------------------------------------- stage 3: out proj + residual
// No LDS, no barrier: A fragments are pre-packed by stage 2, B by stage 0.
__global__ __launch_bounds__(256) void msda_out(const _Float16* __restrict__ attnP,
                                                const _Float16* __restrict__ BoutP,
                                                const float* __restrict__ bout,
                                                const float* __restrict__ query,
                                                float* __restrict__ out) {
    int b     = blockIdx.x >> 10;
    int mtile = blockIdx.x & 1023;
    int q0    = mtile * 16;

    int w = threadIdx.x >> 5, lane = threadIdx.x & 31;
    int col = lane & 15, khalf = lane >> 4;
    int ntile = w;   // 8 waves, 8 N-tiles of Wout's 128 columns

    // Issue all 8 fragment loads (4 A + 4 B), then the WMMA chain.
    const _Float16* ap = attnP + ((size_t)b * 1024 + mtile) * 2048 + (size_t)lane * 16;
    v16h a0 = *reinterpret_cast<const v16h*>(ap);
    v16h a1 = *reinterpret_cast<const v16h*>(ap + 32 * 16);
    v16h a2 = *reinterpret_cast<const v16h*>(ap + 2 * 32 * 16);
    v16h a3 = *reinterpret_cast<const v16h*>(ap + 3 * 32 * 16);

    const _Float16* bp = BoutP + ((size_t)ntile * 4 * 32 + lane) * 16;
    v16h b0 = *reinterpret_cast<const v16h*>(bp);
    v16h b1 = *reinterpret_cast<const v16h*>(bp + 32 * 16);
    v16h b2 = *reinterpret_cast<const v16h*>(bp + 2 * 32 * 16);
    v16h b3 = *reinterpret_cast<const v16h*>(bp + 3 * 32 * 16);

    v8f acc = {};
    acc = WMMA_F16(a0, b0, acc);
    acc = WMMA_F16(a1, b1, acc);
    acc = WMMA_F16(a2, b2, acc);
    acc = WMMA_F16(a3, b3, acc);

    int n = ntile * 16 + col;
    float bb = bout[n];
    size_t rowbase = ((size_t)b * NQ + q0 + khalf * 8) * 128 + n;
    const float* qp = query + rowbase;
    float*       op = out   + rowbase;
#pragma unroll
    for (int i = 0; i < 8; ++i) op[i * 128] = acc[i] + bb + 2.0f * qp[i * 128];
}

// ---------------------------------------------------------------- launcher
extern "C" void kernel_launch(void* const* d_in, const int* in_sizes, int n_in,
                              void* d_out, int out_size, void* d_ws, size_t ws_size,
                              hipStream_t stream) {
    const float* query = (const float*)d_in[0];
    const float* Wv    = (const float*)d_in[1];
    const float* bv    = (const float*)d_in[2];
    const float* Woff  = (const float*)d_in[3];
    const float* boff  = (const float*)d_in[4];
    const float* Wattn = (const float*)d_in[5];
    const float* battn = (const float*)d_in[6];
    const float* Wout  = (const float*)d_in[7];
    const float* bout  = (const float*)d_in[8];
    float* out = (float*)d_out;

    // workspace layout (all offsets 256B aligned)
    char* ws = (char*)d_ws;
    _Float16* BallP  = (_Float16*)(ws);                                    //  57,344 B
    _Float16* BoutP  = (_Float16*)(ws + 57344);                            //  32,768 B
    float*    valueB = (float*)(ws + 90112);                               //  64 MB (b,h,q,16) f32
    float*    offB   = (float*)(ws + 90112 + 67108864ull);                 //  32 MB
    float*    logitB = (float*)(ws + 90112 + 67108864ull + 33554432ull);   //  16 MB
    _Float16* attnP  = (_Float16*)(ws + 90112 + 67108864ull + 33554432ull + 16777216ull); // 32 MB f16

    prep_pack<<<112, 256, 0, stream>>>(Wv, Woff, Wattn, Wout, BallP, BoutP);
    msda_proj<<<8192, 256, 0, stream>>>(query, BallP, bv, boff, battn, valueB, offB, logitB);
    msda_sample<<<4096, 256, 0, stream>>>(valueB, offB, logitB, attnP);
    msda_out<<<8192, 256, 0, stream>>>(attnP, BoutP, bout, query, out);
    (void)in_sizes; (void)n_in; (void)out_size; (void)ws_size;
}
